// SetAbstraction_64622077935821
// MI455X (gfx1250) — matrix operand order repaired and checked by reading, not compiled, not run on hardware
//
#include <hip/hip_runtime.h>

// PointNet++ Set Abstraction for MI455X (gfx1250, wave32, WMMA + TDM/async-LDS).
// Pipeline: FPS -> KNN -> gather/concat(f16) -> [WMMA GEMM -> BN stats -> BN+ReLU] x3 -> maxpool.

typedef _Float16 half_t;
typedef __attribute__((ext_vector_type(16))) _Float16 v16h;
typedef __attribute__((ext_vector_type(8)))  _Float16 v8h;
typedef __attribute__((ext_vector_type(8)))  float    v8f;

#define BB 16
#define NN 4096
#define SS 1024
#define KK 32
#define CIN 64
#define KD1 96        // 67 channels padded to 3 WMMA k-steps
#define KROWS1 (CIN + 3)
#define MROWS (BB * SS * KK)   // 524288 GEMM rows

// ---- gfx1250 async / TDM capability probes (device pass only) --------------
#if defined(__gfx1250__) && __has_builtin(__builtin_amdgcn_tensor_load_to_lds) && \
    __has_builtin(__builtin_amdgcn_s_wait_tensorcnt)
#define HAVE_TDM 1
#endif
#if defined(__gfx1250__) && __has_builtin(__builtin_amdgcn_global_load_async_to_lds_b128) && \
    __has_builtin(__builtin_amdgcn_s_wait_asynccnt)
#define HAVE_ASYNC 1
#endif

#if defined(HAVE_TDM) || defined(HAVE_ASYNC)
typedef unsigned int u32;
typedef __attribute__((ext_vector_type(4))) unsigned int u32x4;
typedef __attribute__((ext_vector_type(4))) int i32x4;
typedef __attribute__((ext_vector_type(8))) int i32x8;
#endif
#ifdef HAVE_ASYNC
// Parameter type per hipcc diagnostic: pointer to 4 x i32 vector (generic AS).
typedef int v4i_b128 __attribute__((vector_size(16)));
#endif

// ---------------------------------------------------------------------------
// 1) Farthest point sampling: one workgroup per batch, xyz cached in LDS.
//    Writes new_xyz [B,S,3] directly into d_out.
// ---------------------------------------------------------------------------
__global__ __launch_bounds__(256) void fps_kernel(const float* __restrict__ xyz,
                                                  float* __restrict__ new_xyz) {
  __shared__ float sx[NN], sy[NN], sz[NN];
  __shared__ float rd[256];
  __shared__ int   ri[256];
  __shared__ float cpt[3];
  const int b = blockIdx.x, tid = threadIdx.x;
  const float* p = xyz + (size_t)b * NN * 3;
  for (int i = tid; i < NN; i += 256) {
    sx[i] = p[3 * i]; sy[i] = p[3 * i + 1]; sz[i] = p[3 * i + 2];
  }
  float dmin[16];
#pragma unroll
  for (int i = 0; i < 16; ++i) dmin[i] = 1e10f;
  __syncthreads();

  int far = 0;
  float* outb = new_xyz + (size_t)b * SS * 3;
  for (int s = 0; s < SS; ++s) {
    if (tid == 0) {
      float x = sx[far], y = sy[far], z = sz[far];
      outb[3 * s] = x; outb[3 * s + 1] = y; outb[3 * s + 2] = z;
      cpt[0] = x; cpt[1] = y; cpt[2] = z;
    }
    __syncthreads();
    const float bx = cpt[0], by = cpt[1], bz = cpt[2];
    float best = -1.0f; int bidx = tid;
#pragma unroll
    for (int i = 0; i < 16; ++i) {
      int n = i * 256 + tid;
      float dx = sx[n] - bx, dy = sy[n] - by, dz = sz[n] - bz;
      float d = dx * dx + dy * dy + dz * dz;
      float dm = fminf(dmin[i], d);
      dmin[i] = dm;
      if (dm > best) { best = dm; bidx = n; }
    }
    rd[tid] = best; ri[tid] = bidx;
    __syncthreads();
    for (int st = 128; st > 0; st >>= 1) {
      if (tid < st && rd[tid + st] > rd[tid]) {
        rd[tid] = rd[tid + st]; ri[tid] = ri[tid + st];
      }
      __syncthreads();
    }
    far = ri[0];
    __syncthreads();
  }
}

// ---------------------------------------------------------------------------
// 2) KNN: one thread per query; candidate xyz streamed through LDS tiles via
//    GLOBAL_LOAD_ASYNC_TO_LDS_B128 (ASYNCcnt) when available.
// ---------------------------------------------------------------------------
__global__ __launch_bounds__(256) void knn_kernel(const float* __restrict__ xyz,
                                                  const float* __restrict__ new_xyz,
                                                  int* __restrict__ knn_idx) {
  __shared__ float txyz[3072];   // 1024 points, interleaved x,y,z (12 KB)
  const int b = blockIdx.x >> 2;
  const int s = ((blockIdx.x & 3) << 8) | threadIdx.x;
  const float* q = new_xyz + ((size_t)b * SS + s) * 3;
  const float qx = q[0], qy = q[1], qz = q[2];
  float kd[KK]; int ki[KK];
#pragma unroll
  for (int i = 0; i < KK; ++i) { kd[i] = 3.0e38f; ki[i] = 0; }
  const float* p = xyz + (size_t)b * NN * 3;
  for (int tile = 0; tile < 4; ++tile) {
    __syncthreads();
#ifdef HAVE_ASYNC
    {
      const float* gp = p + tile * 3072 + threadIdx.x * 4;
      float* lp = &txyz[threadIdx.x * 4];
#pragma unroll
      for (int it = 0; it < 3; ++it)
        __builtin_amdgcn_global_load_async_to_lds_b128(
            (v4i_b128*)(gp + it * 1024), (v4i_b128*)(lp + it * 1024), 0, 0);
      __builtin_amdgcn_s_wait_asynccnt(0);
    }
#else
    {
      const float4* gp = (const float4*)(p + tile * 3072);
      for (int i = threadIdx.x; i < 768; i += 256) ((float4*)txyz)[i] = gp[i];
    }
#endif
    if (tile < 3) __builtin_prefetch(p + 3 * ((tile + 1) << 10) + 3 * threadIdx.x, 0, 1);
    __syncthreads();
    for (int i = 0; i < 1024; ++i) {
      float dx = txyz[3 * i] - qx, dy = txyz[3 * i + 1] - qy, dz = txyz[3 * i + 2] - qz;
      float d = dx * dx + dy * dy + dz * dz;
      if (d < kd[KK - 1]) {
        int j = KK - 1;
        while (j > 0 && kd[j - 1] > d) { kd[j] = kd[j - 1]; ki[j] = ki[j - 1]; --j; }
        kd[j] = d; ki[j] = (tile << 10) | i;
      }
    }
  }
  int* o = knn_idx + ((size_t)b * SS + s) * KK;
#pragma unroll
  for (int i = 0; i < KK; ++i) o[i] = ki[i];
}

// ---------------------------------------------------------------------------
// 3) Transpose features [B,C,N] -> [B,N,C] for contiguous row gathers.
// ---------------------------------------------------------------------------
__global__ void feat_transpose_kernel(const float* __restrict__ f, float* __restrict__ ft) {
  size_t i = (size_t)blockIdx.x * 256 + threadIdx.x;   // B*N*C, c fastest
  int c = (int)(i & (CIN - 1));
  int n = (int)((i >> 6) & (NN - 1));
  int b = (int)(i >> 18);
  ft[i] = f[((size_t)b * CIN + c) * NN + n];
}

// ---------------------------------------------------------------------------
// 4) Gather + concat -> padded f16 activation matrix X [MROWS, 96].
// ---------------------------------------------------------------------------
__global__ void gather_kernel(const float* __restrict__ xyz, const float* __restrict__ new_xyz,
                              const float* __restrict__ featT, const int* __restrict__ knn_idx,
                              half_t* __restrict__ X) {
  size_t row = (size_t)blockIdx.x * 256 + threadIdx.x;  // < MROWS
  int s = (int)((row >> 5) & (SS - 1));
  int b = (int)(row >> 15);
  int n = knn_idx[row];
  const float* pc = xyz + ((size_t)b * NN + n) * 3;
  const float* cc = new_xyz + ((size_t)b * SS + s) * 3;
  half_t* xr = X + row * (size_t)KD1;
  xr[0] = (half_t)(pc[0] - cc[0]);
  xr[1] = (half_t)(pc[1] - cc[1]);
  xr[2] = (half_t)(pc[2] - cc[2]);
  const float* f = featT + ((size_t)b * NN + n) * CIN;
#pragma unroll 8
  for (int c = 0; c < CIN; ++c) xr[3 + c] = (half_t)f[c];
#pragma unroll
  for (int c = 3 + CIN; c < KD1; ++c) xr[c] = (half_t)0.0f;
}

// ---------------------------------------------------------------------------
// 5) Pack weights into per-lane WMMA B-fragment order (ISA 7.12.2):
//    elem h<8 holds K=kbase+8*(lane/16)+h, h>=8 holds K=kbase+16+8*(lane/16)+(h-8); col = j*16 + lane%16.
// ---------------------------------------------------------------------------
__global__ void pack_w_kernel(const float* __restrict__ w, half_t* __restrict__ wp,
                              int krows, int ktiles, int ntiles, int nout) {
  int tid = blockIdx.x * 256 + threadIdx.x;
  if (tid >= ktiles * ntiles * 32) return;
  int lane = tid & 31, frag = tid >> 5;
  int j = frag % ntiles, t = frag / ntiles;
  int col = j * 16 + (lane & 15);
  int kb = t * 32 + ((lane >> 4) << 4);
  half_t* o = wp + (size_t)tid * 16;
#pragma unroll
  for (int h = 0; h < 16; ++h) {
    int k = t * 32 + ((lane >> 4) << 3) + (h & 7) + ((h >> 3) << 4);
    o[h] = (half_t)((k < krows) ? w[(size_t)k * nout + col] : 0.0f);
  }
}

// ---------------------------------------------------------------------------
// 6) WMMA GEMM: Y[M,NOUT] = X[M,KD] * W[KD,NOUT]. 8 waves/block, each wave
//    owns 16 rows x NOUT cols. Packed weights staged block-wide into LDS by
//    the Tensor Data Mover (TENSORcnt) when available.
// ---------------------------------------------------------------------------
template <int KD, int NOUT>
__global__ __launch_bounds__(256) void gemm_wmma_kernel(const half_t* __restrict__ X,
                                                        const half_t* __restrict__ Wp,
                                                        float* __restrict__ Y) {
  constexpr int KT = KD / 32, NT = NOUT / 16;
  constexpr int WELEMS = KT * NT * 512;          // packed halves
  const int lane = threadIdx.x & 31;
  const int wv = threadIdx.x >> 5;

#ifdef HAVE_TDM
  __shared__ half_t wsmem[WELEMS];
  if (wv == 0) {
    // 1-D D# descriptor: count=1, type=2(image), data_size=8B, tile==tensor.
    const unsigned long long ga = (unsigned long long)(uintptr_t)Wp;
    const unsigned lds_off = (unsigned)(unsigned long long)(uintptr_t)(&wsmem[0]); // low 32b = LDS offset
    constexpr unsigned NE8 = (unsigned)(WELEMS / 4);   // 8-byte elements
    u32x4 g0;
    g0[0] = 1u;                                        // count=1 (valid user desc)
    g0[1] = lds_off;                                   // lds_addr
    g0[2] = (u32)(ga & 0xffffffffu);                   // global_addr[31:0]
    g0[3] = (u32)((ga >> 32) & 0x01ffffffu) | 0x80000000u; // global_addr[56:32] | type=2
    i32x8 g1;
    g1[0] = (int)(3u << 16);                           // data_size = 3 (8 bytes)
    g1[1] = (int)((NE8 & 0xffffu) << 16);              // tensor_dim0[15:0]
    g1[2] = (int)((NE8 >> 16) | (1u << 16));           // tensor_dim0[31:16], tensor_dim1=1
    g1[3] = (int)((NE8 & 0xffffu) << 16);              // tile_dim0
    g1[4] = 1;                                         // tile_dim1 = 1
    g1[5] = (int)NE8;                                  // tensor_dim0_stride lo
    g1[6] = 0; g1[7] = 0;
    i32x4 z4; z4[0] = 0; z4[1] = 0; z4[2] = 0; z4[3] = 0;
#if __clang_major__ >= 23
    i32x8 z8; for (int i = 0; i < 8; ++i) z8[i] = 0;
    __builtin_amdgcn_tensor_load_to_lds(g0, g1, z4, z4, z8, 0);
#else
    __builtin_amdgcn_tensor_load_to_lds(g0, g1, z4, z4, 0);
#endif
    __builtin_amdgcn_s_wait_tensorcnt(0);
  }
  __syncthreads();
  const half_t* wbase = wsmem;
#else
  const half_t* wbase = Wp;
#endif

  const int row0 = (blockIdx.x * 8 + wv) << 4;
  const int hsel = (lane < 16) ? 0 : 8;    // A-fragment K sub-select per half-wave
  const half_t* ap = X + ((size_t)row0 + (lane & 15)) * KD;

  v8f acc[NT];
#pragma unroll
  for (int j = 0; j < NT; ++j)
#pragma unroll
    for (int r = 0; r < 8; ++r) acc[j][r] = 0.0f;

#pragma unroll
  for (int t = 0; t < KT; ++t) {
    v8h lo = *(const v8h*)(ap + t * 32 + hsel);
    v8h hi = *(const v8h*)(ap + t * 32 + 16 + hsel);
    v16h a;
#pragma unroll
    for (int i = 0; i < 8; ++i) { a[i] = lo[i]; a[i + 8] = hi[i]; }
#pragma unroll
    for (int j = 0; j < NT; ++j) {
      const v8h* bp = (const v8h*)(wbase + (((size_t)t * NT + j) * 32 + lane) * 16);
      v8h bl = bp[0], bh = bp[1];
      v16h bv;
#pragma unroll
      for (int i = 0; i < 8; ++i) { bv[i] = bl[i]; bv[i + 8] = bh[i]; }
      acc[j] = __builtin_amdgcn_wmma_f32_16x16x32_f16(
          /*neg_a=*/false, a, /*neg_b=*/false, bv,
          /*c_mod=*/(short)0, acc[j], /*reuse_a=*/false, /*reuse_b=*/false);
    }
  }

  // D layout: VGPR r -> M = row0 + r + (lane<16 ? 0 : 8), N = j*16 + lane%16
  const int mo = row0 + ((lane < 16) ? 0 : 8);
  const int col = lane & 15;
#pragma unroll
  for (int j = 0; j < NT; ++j) {
    float* yp = Y + (size_t)mo * NOUT + j * 16 + col;
#pragma unroll
    for (int r = 0; r < 8; ++r) yp[(size_t)r * NOUT] = acc[j][r];
  }
}

// ---------------------------------------------------------------------------
// 7) BN batch statistics (train-mode): per-channel sum & sumsq.
// ---------------------------------------------------------------------------
__global__ void zero_stats_kernel(float* __restrict__ s) { s[threadIdx.x] = 0.0f; }

template <int NC>
__global__ void bn_stats_kernel(const float* __restrict__ Y, float* __restrict__ stats,
                                int rowsPerBlock) {
  const int c = threadIdx.x;   // blockDim.x == NC -> coalesced column reads
  size_t r0 = (size_t)blockIdx.x * rowsPerBlock;
  float s = 0.0f, q = 0.0f;
  for (int r = 0; r < rowsPerBlock; ++r) {
    float v = Y[(r0 + r) * NC + c];
    s += v; q += v * v;
  }
  atomicAdd(&stats[c], s);
  atomicAdd(&stats[NC + c], q);
}

// ---------------------------------------------------------------------------
// 8) BN apply + ReLU + f16 pack for next GEMM (layers 1 & 2).
// ---------------------------------------------------------------------------
template <int NC>
__global__ void bn_apply_pack_kernel(const float* __restrict__ Y, const float* __restrict__ stats,
                                     const float* __restrict__ g, const float* __restrict__ bta,
                                     half_t* __restrict__ Xn) {
  size_t i = (size_t)blockIdx.x * 256 + threadIdx.x;
  int c = (int)(i & (NC - 1));
  const float inv = 1.0f / (float)MROWS;
  float mean = stats[c] * inv;
  float var = stats[NC + c] * inv - mean * mean;
  float v = (Y[i] - mean) * rsqrtf(var + 1e-5f) * g[c] + bta[c];
  Xn[i] = (half_t)fmaxf(v, 0.0f);
}

// ---------------------------------------------------------------------------
// 9) Layer-3 BN + ReLU + max over K neighbors + transposed store [B,128,S].
// ---------------------------------------------------------------------------
__global__ void bn_maxpool_kernel(const float* __restrict__ Y, const float* __restrict__ stats,
                                  const float* __restrict__ g, const float* __restrict__ bta,
                                  float* __restrict__ out) {
  size_t i = (size_t)blockIdx.x * 256 + threadIdx.x;   // B*S*128, c fastest
  int c = (int)(i & 127);
  size_t bs = i >> 7;
  const float inv = 1.0f / (float)MROWS;
  float mean = stats[c] * inv;
  float var = stats[128 + c] * inv - mean * mean;
  float sc = rsqrtf(var + 1e-5f) * g[c];
  float sh = bta[c] - mean * sc;
  const float* yp = Y + bs * (size_t)(KK * 128) + c;
  float m = 0.0f;   // ReLU floor: max(relu(v)) == max(0, max(v))
#pragma unroll 8
  for (int k = 0; k < KK; ++k) m = fmaxf(m, yp[(size_t)k * 128] * sc + sh);
  int b = (int)(bs >> 10);
  int s = (int)(bs & (SS - 1));
  out[(size_t)b * 128 * SS + (size_t)c * SS + s] = m;
}

// ---------------------------------------------------------------------------
extern "C" void kernel_launch(void* const* d_in, const int* in_sizes, int n_in,
                              void* d_out, int out_size, void* d_ws, size_t ws_size,
                              hipStream_t stream) {
  const float* xyz  = (const float*)d_in[0];
  const float* feat = (const float*)d_in[1];
  const float* w1 = (const float*)d_in[2];
  const float* g1 = (const float*)d_in[3];
  const float* b1 = (const float*)d_in[4];
  const float* w2 = (const float*)d_in[5];
  const float* g2 = (const float*)d_in[6];
  const float* b2 = (const float*)d_in[7];
  const float* w3 = (const float*)d_in[8];
  const float* g3 = (const float*)d_in[9];
  const float* b3 = (const float*)d_in[10];
  (void)in_sizes; (void)n_in; (void)out_size; (void)ws_size;

  float* out_xyz  = (float*)d_out;                       // [B,S,3]
  float* out_feat = (float*)d_out + (size_t)BB * SS * 3; // [B,128,S]

  char* ws = (char*)d_ws;
  size_t off = 0;
  int*    knn_idx = (int*)(ws + off);    off += (size_t)BB * SS * KK * 4;        // 2 MB
  float*  featT   = (float*)(ws + off);  off += (size_t)BB * NN * CIN * 4;       // 16 MB
  half_t* X       = (half_t*)(ws + off); off += (size_t)MROWS * KD1 * 2;         // 96 MB (reused all layers)
  float*  Y       = (float*)(ws + off);  off += (size_t)MROWS * 128 * 4;         // 256 MB (reused all layers)
  half_t* wp1     = (half_t*)(ws + off); off += 32768;
  half_t* wp2     = (half_t*)(ws + off); off += 32768;
  half_t* wp3     = (half_t*)(ws + off); off += 32768;
  float*  stats   = (float*)(ws + off);                                          // 256 floats

  // Sampling + grouping
  fps_kernel<<<BB, 256, 0, stream>>>(xyz, out_xyz);
  knn_kernel<<<BB * 4, 256, 0, stream>>>(xyz, out_xyz, knn_idx);
  feat_transpose_kernel<<<(BB * NN * CIN) / 256, 256, 0, stream>>>(feat, featT);
  gather_kernel<<<MROWS / 256, 256, 0, stream>>>(xyz, out_xyz, featT, knn_idx, X);

  // Layer 1: [M,96] x [96,64]
  pack_w_kernel<<<2, 256, 0, stream>>>(w1, wp1, KROWS1, 3, 4, 64);
  zero_stats_kernel<<<1, 256, 0, stream>>>(stats);
  gemm_wmma_kernel<96, 64><<<MROWS / 128, 256, 0, stream>>>(X, wp1, Y);
  bn_stats_kernel<64><<<MROWS / 512, 64, 0, stream>>>(Y, stats, 512);
  bn_apply_pack_kernel<64><<<(MROWS / 256) * 64, 256, 0, stream>>>(Y, stats, g1, b1, X);

  // Layer 2: [M,64] x [64,64]
  pack_w_kernel<<<1, 256, 0, stream>>>(w2, wp2, 64, 2, 4, 64);
  zero_stats_kernel<<<1, 256, 0, stream>>>(stats);
  gemm_wmma_kernel<64, 64><<<MROWS / 128, 256, 0, stream>>>(X, wp2, Y);
  bn_stats_kernel<64><<<MROWS / 512, 64, 0, stream>>>(Y, stats, 512);
  bn_apply_pack_kernel<64><<<(MROWS / 256) * 64, 256, 0, stream>>>(Y, stats, g2, b2, X);

  // Layer 3: [M,64] x [64,128] + maxpool
  pack_w_kernel<<<2, 256, 0, stream>>>(w3, wp3, 64, 2, 8, 128);
  zero_stats_kernel<<<1, 256, 0, stream>>>(stats);
  gemm_wmma_kernel<64, 128><<<MROWS / 128, 256, 0, stream>>>(X, wp3, Y);
  bn_stats_kernel<128><<<MROWS / 512, 128, 0, stream>>>(Y, stats, 512);
  bn_maxpool_kernel<<<(BB * SS * 128) / 256, 256, 0, stream>>>(Y, stats, g3, b3, out_feat);
}
